// GAT_41008347742283
// MI455X (gfx1250) — matrix operand order repaired and checked by reading, not compiled, not run on hardware
//
#include <hip/hip_runtime.h>
#include <hip/hip_bf16.h>

// ---------------- problem constants (match reference) ----------------
#define NODES 20000
#define EDGES 320000
#define ETOT  (EDGES + NODES)   // self loops appended
#define SLOPE 0.2f

typedef __bf16 bf16_t;
typedef bf16_t v16bf __attribute__((ext_vector_type(16)));
typedef bf16_t v8bf  __attribute__((ext_vector_type(8)));
typedef float  v8f   __attribute__((ext_vector_type(8)));

// ---------------- order-preserving float<->uint for atomic max ----------------
__device__ __forceinline__ unsigned f2ord(float f) {
    unsigned b = __float_as_uint(f);
    return (b >> 31) ? ~b : (b | 0x80000000u);
}
__device__ __forceinline__ float ord2f(unsigned u) {
    return __uint_as_float((u >> 31) ? (u & 0x7FFFFFFFu) : ~u);
}
#define ORD_NEG_INF 0x007FFFFFu   // f2ord(-inf)

// ---------------- edge list with self loops ----------------
__global__ void build_edges(const int* __restrict__ ei,
                            int* __restrict__ srcF, int* __restrict__ dstF) {
    int t = blockIdx.x * blockDim.x + threadIdx.x;
    if (t >= ETOT) return;
    if (t < EDGES) { srcF[t] = ei[t]; dstF[t] = ei[EDGES + t]; }
    else           { srcF[t] = t - EDGES; dstF[t] = t - EDGES; }
}

// ---------------- f32 -> bf16 conversions ----------------
__global__ void cvt_bf16(const float* __restrict__ in, bf16_t* __restrict__ out, int n) {
    int t = blockIdx.x * blockDim.x + threadIdx.x;
    if (t < n) out[t] = (bf16_t)in[t];
}

__global__ void cvt_w(const float* __restrict__ wl, const float* __restrict__ wr,
                      bf16_t* __restrict__ out, int n) {
    int t = blockIdx.x * blockDim.x + threadIdx.x;
    if (t < n) { out[t] = (bf16_t)wl[t]; out[n + t] = (bf16_t)wr[t]; }
}

// ---------------- per-layer scratch init ----------------
__global__ void init_layer(float* __restrict__ acc, unsigned* __restrict__ nmax,
                           float* __restrict__ den, int nacc, int nh) {
    int t = blockIdx.x * blockDim.x + threadIdx.x;
    if (t < nacc) acc[t] = 0.f;
    if (t < nh) { nmax[t] = ORD_NEG_INF; den[t] = 0.f; }
}

// ---------------- WMMA bf16 GEMM ----------------
// Y[N x OC] = X[N x IC] * W[OC x IC]^T + bias
// One wave computes a 16 x (16*NT) tile. Fragments are DOUBLE-BUFFERED across
// the K loop: iteration k issues the loads for k into the 'next' set before the
// WMMAs consume the 'current' set, so matrix ops overlap a full iteration of
// in-flight loads instead of draining loadcnt to zero.
// Fragment layouts per CDNA5 ISA 7.12.2 (A 16x32 bf16, B 32x16 bf16, C/D 16x16 f32).
template <int NT>
__global__ void __launch_bounds__(256)
gemm_bf16_wmma(const bf16_t* __restrict__ X, const bf16_t* __restrict__ W,
               const float* __restrict__ bias, float* __restrict__ Y,
               int IC, int OC, int groupsN, int wavesTotal) {
    int wave = blockIdx.x * (blockDim.x >> 5) + (threadIdx.x >> 5);
    if (wave >= wavesTotal) return;
    int tm = wave / groupsN;
    int tg = wave - tm * groupsN;       // group of NT consecutive 16-wide N tiles
    int lane = threadIdx.x & 31;
    int half = lane >> 4;               // 0: lanes 0-15, 1: lanes 16-31
    int l15  = lane & 15;

    // A: row = tm*16 + l15 ; K chunks {half*8..+7} in v0-3, {16+half*8..+7} in v4-7
    const bf16_t* ap = X + (size_t)(tm * 16 + l15) * IC + half * 8;
    // B: B[k][n] = W[n][k]; lane carries column n=l15, K chunks {half*8..+7}, {16+half*8..+7}
    const bf16_t* bp[NT];
    v8f c[NT];
#pragma unroll
    for (int nt = 0; nt < NT; ++nt) {
        bp[nt] = W + (size_t)((tg * NT + nt) * 16 + l15) * IC + half * 8;
        c[nt] = (v8f){0.f, 0.f, 0.f, 0.f, 0.f, 0.f, 0.f, 0.f};
    }

    // ---- preload K-step 0 ----
    v8bf a0 = *(const v8bf*)(ap);            // K = half*8 + 0..7
    v8bf a1 = *(const v8bf*)(ap + 16);       // K = 16 + half*8 + 0..7
    v8bf b0[NT], b1[NT];
#pragma unroll
    for (int nt = 0; nt < NT; ++nt) {
        b0[nt] = *(const v8bf*)(bp[nt]);     // K = half*8 + 0..7
        b1[nt] = *(const v8bf*)(bp[nt] + 16); // K = 16 + half*8 + 0..7
    }

    // ---- steady state: issue loads for k, compute on k-32 ----
    for (int k = 32; k <= IC - 32; k += 32) {
        v8bf na0 = *(const v8bf*)(ap + k);
        v8bf na1 = *(const v8bf*)(ap + k + 16);
        v8bf nb0[NT], nb1[NT];
#pragma unroll
        for (int nt = 0; nt < NT; ++nt) {
            nb0[nt] = *(const v8bf*)(bp[nt] + k);
            nb1[nt] = *(const v8bf*)(bp[nt] + k + 16);
        }

        v16bf A;
#pragma unroll
        for (int i = 0; i < 8; ++i) { A[i] = a0[i]; A[i + 8] = a1[i]; }
#pragma unroll
        for (int nt = 0; nt < NT; ++nt) {
            v16bf B;
#pragma unroll
            for (int i = 0; i < 8; ++i) { B[i] = b0[nt][i]; B[i + 8] = b1[nt][i]; }
            c[nt] = __builtin_amdgcn_wmma_f32_16x16x32_bf16(
                        false, A, false, B, (short)0, c[nt], false, false);
        }

        a0 = na0; a1 = na1;
#pragma unroll
        for (int nt = 0; nt < NT; ++nt) { b0[nt] = nb0[nt]; b1[nt] = nb1[nt]; }
    }

    // ---- epilogue K-step ----
    {
        v16bf A;
#pragma unroll
        for (int i = 0; i < 8; ++i) { A[i] = a0[i]; A[i + 8] = a1[i]; }
#pragma unroll
        for (int nt = 0; nt < NT; ++nt) {
            v16bf B;
#pragma unroll
            for (int i = 0; i < 8; ++i) { B[i] = b0[nt][i]; B[i + 8] = b1[nt][i]; }
            c[nt] = __builtin_amdgcn_wmma_f32_16x16x32_bf16(
                        false, A, false, B, (short)0, c[nt], false, false);
        }
    }

#pragma unroll
    for (int nt = 0; nt < NT; ++nt) {
        int col = (tg * NT + nt) * 16 + l15;
        float bv = bias[col];
#pragma unroll
        for (int r = 0; r < 8; ++r) {
            int row = tm * 16 + half * 8 + r;   // C: lanes<16 -> M=r, else M=8+r
            Y[(size_t)row * OC + col] = c[nt][r] + bv;
        }
    }
}

// ---------------- per-edge attention logits + segment max (fused) ----------------
// wave per edge; H=4: 8 lanes/head (channels contiguous); H=1/oc=16: lanes 0-15
__global__ void __launch_bounds__(256)
edge_logits(const float* __restrict__ xl, const float* __restrict__ xr,
            const float* __restrict__ att,
            const int* __restrict__ srcF, const int* __restrict__ dstF,
            float* __restrict__ logits, unsigned* __restrict__ nmax,
            int heads, int hoc) {
    int e = blockIdx.x * (blockDim.x >> 5) + (threadIdx.x >> 5);
    if (e >= ETOT) return;
    int lane = threadIdx.x & 31;
    if (lane == 0 && e + 256 < ETOT) {
        __builtin_prefetch((const void*)(srcF + e + 256), 0, 0);  // global_prefetch_b8
        __builtin_prefetch((const void*)(dstF + e + 256), 0, 0);
    }
    int s = srcF[e], d = dstF[e];

    if (hoc == 256) {
        const float4* pl = (const float4*)(xl + (size_t)s * 256) + lane * 2;
        const float4* pr = (const float4*)(xr + (size_t)d * 256) + lane * 2;
        const float4* pa = (const float4*)att + lane * 2;
        float acc = 0.f;
#pragma unroll
        for (int q = 0; q < 2; ++q) {
            float4 vl = pl[q], vr = pr[q], va = pa[q];
            float v;
            v = vl.x + vr.x; v = v > 0.f ? v : SLOPE * v; acc += v * va.x;
            v = vl.y + vr.y; v = v > 0.f ? v : SLOPE * v; acc += v * va.y;
            v = vl.z + vr.z; v = v > 0.f ? v : SLOPE * v; acc += v * va.z;
            v = vl.w + vr.w; v = v > 0.f ? v : SLOPE * v; acc += v * va.w;
        }
        acc += __shfl_xor(acc, 1, 32);
        acc += __shfl_xor(acc, 2, 32);
        acc += __shfl_xor(acc, 4, 32);
        if ((lane & 7) == 0) {
            int h = lane >> 3;
            logits[(size_t)e * 4 + h] = acc;
            atomicMax(&nmax[(size_t)d * 4 + h], f2ord(acc));  // global_atomic_max_u32
        }
    } else {  // heads==1, hoc==16
        float v = 0.f;
        if (lane < 16) {
            float t = xl[(size_t)s * 16 + lane] + xr[(size_t)d * 16 + lane];
            t = t > 0.f ? t : SLOPE * t;
            v = t * att[lane];
        }
        v += __shfl_xor(v, 1, 32);
        v += __shfl_xor(v, 2, 32);
        v += __shfl_xor(v, 4, 32);
        v += __shfl_xor(v, 8, 32);
        if (lane == 0) {
            logits[e] = v;
            atomicMax(&nmax[d], f2ord(v));
        }
    }
}

// ---------------- exp(logit - max) and segment-sum of denominators ----------------
__global__ void edge_expsum(const int* __restrict__ dstF, float* __restrict__ logits,
                            const unsigned* __restrict__ nmax, float* __restrict__ den,
                            int heads) {
    int t = blockIdx.x * blockDim.x + threadIdx.x;
    if (t >= ETOT * heads) return;
    int e, h;
    if (heads == 4) { e = t >> 2; h = t & 3; }
    else            { e = t;      h = 0;     }
    int d = dstF[e];
    float m = ord2f(nmax[(size_t)d * heads + h]);
    float ex = __expf(logits[t] - m);
    logits[t] = ex;                                   // reuse buffer as ex
    atomicAdd(&den[(size_t)d * heads + h], ex);
}

// ---------------- weighted scatter-add aggregation ----------------
__global__ void __launch_bounds__(256)
edge_aggregate(const float* __restrict__ xl, const float* __restrict__ ex,
               const float* __restrict__ den,
               const int* __restrict__ srcF, const int* __restrict__ dstF,
               float* __restrict__ acc, int heads, int hoc) {
    int e = blockIdx.x * (blockDim.x >> 5) + (threadIdx.x >> 5);
    if (e >= ETOT) return;
    int lane = threadIdx.x & 31;
    if (lane == 0 && e + 256 < ETOT) {
        __builtin_prefetch((const void*)(srcF + e + 256), 0, 0);
    }
    int s = srcF[e], d = dstF[e];

    if (hoc == 256) {
        int h = lane >> 3;
        float alpha = ex[(size_t)e * 4 + h] / den[(size_t)d * 4 + h];
        const float4* pl = (const float4*)(xl + (size_t)s * 256) + lane * 2;
        float* pd = acc + (size_t)d * 256 + lane * 8;
#pragma unroll
        for (int q = 0; q < 2; ++q) {
            float4 v = pl[q];
            atomicAdd(pd + q * 4 + 0, alpha * v.x);
            atomicAdd(pd + q * 4 + 1, alpha * v.y);
            atomicAdd(pd + q * 4 + 2, alpha * v.z);
            atomicAdd(pd + q * 4 + 3, alpha * v.w);
        }
    } else {
        if (lane < 16) {
            float alpha = ex[e] / den[d];
            atomicAdd(acc + (size_t)d * 16 + lane, alpha * xl[(size_t)s * 16 + lane]);
        }
    }
}

// ---------------- bias + ReLU epilogue, fused with bf16 convert for next layer ----------------
__global__ void epilogue_relu_bf16(const float* __restrict__ acc, const float* __restrict__ bo,
                                   bf16_t* __restrict__ out, int n, int cmask) {
    int t = blockIdx.x * blockDim.x + threadIdx.x;
    if (t >= n) return;
    float v = acc[t] + bo[t & cmask];
    out[t] = (bf16_t)fmaxf(v, 0.f);
}

// ---------------- final layer epilogue: bias only, f32 output ----------------
__global__ void epilogue_out(const float* __restrict__ acc, const float* __restrict__ bo,
                             float* __restrict__ out, int n, int cmask) {
    int t = blockIdx.x * blockDim.x + threadIdx.x;
    if (t >= n) return;
    out[t] = acc[t] + bo[t & cmask];
}

// =====================================================================
extern "C" void kernel_launch(void* const* d_in, const int* in_sizes, int n_in,
                              void* d_out, int out_size, void* d_ws, size_t ws_size,
                              hipStream_t stream) {
    (void)in_sizes; (void)n_in; (void)out_size; (void)ws_size;

    const float* x  = (const float*)d_in[0];
    const int*   ei = (const int*)  d_in[1];
    const float* wl[4] = {(const float*)d_in[2],  (const float*)d_in[8],
                          (const float*)d_in[14], (const float*)d_in[20]};
    const float* bl[4] = {(const float*)d_in[3],  (const float*)d_in[9],
                          (const float*)d_in[15], (const float*)d_in[21]};
    const float* wr[4] = {(const float*)d_in[4],  (const float*)d_in[10],
                          (const float*)d_in[16], (const float*)d_in[22]};
    const float* br[4] = {(const float*)d_in[5],  (const float*)d_in[11],
                          (const float*)d_in[17], (const float*)d_in[23]};
    const float* at[4] = {(const float*)d_in[6],  (const float*)d_in[12],
                          (const float*)d_in[18], (const float*)d_in[24]};
    const float* bo[4] = {(const float*)d_in[7],  (const float*)d_in[13],
                          (const float*)d_in[19], (const float*)d_in[25]};

    const int IC [4] = {128, 256, 256, 256};
    const int HOC[4] = {256, 256, 256, 16};
    const int HDS[4] = {4, 4, 4, 1};

    // -------- workspace carve (256B aligned) --------
    char* wp = (char*)d_ws;
    auto carve = [&](size_t bytes) -> char* {
        char* p = wp;
        wp += (bytes + 255) & ~(size_t)255;
        return p;
    };
    const size_t NB = (size_t)NODES * 256;
    float*    xl    = (float*)   carve(NB * 4);
    float*    xr    = (float*)   carve(NB * 4);
    float*    accb  = (float*)   carve(NB * 4);
    float*    logit = (float*)   carve((size_t)ETOT * 4 * 4);
    unsigned* nmax  = (unsigned*)carve((size_t)NODES * 4 * 4);
    float*    den   = (float*)   carve((size_t)NODES * 4 * 4);
    int*      srcF  = (int*)     carve((size_t)ETOT * 4);
    int*      dstF  = (int*)     carve((size_t)ETOT * 4);
    bf16_t*   xbf   = (bf16_t*)  carve(NB * 2);
    bf16_t*   wbf   = (bf16_t*)  carve((size_t)2 * 256 * 256 * 2);

    build_edges<<<(ETOT + 255) / 256, 256, 0, stream>>>(ei, srcF, dstF);
    cvt_bf16<<<(NODES * 128 + 255) / 256, 256, 0, stream>>>(x, xbf, NODES * 128);

    for (int li = 0; li < 4; ++li) {
        const int ic = IC[li], hoc = HOC[li], heads = HDS[li];
        const int nW = hoc * ic;
        const int nA = NODES * hoc;

        cvt_w<<<(nW + 255) / 256, 256, 0, stream>>>(wl[li], wr[li], wbf, nW);
        init_layer<<<(nA + 255) / 256, 256, 0, stream>>>(accb, nmax, den, nA, NODES * heads);

        if (hoc == 256) {                       // NT=4: 16x64 tile per wave
            const int groupsN = 256 / 64;
            const int waves   = (NODES / 16) * groupsN;
            const int gblk    = (waves + 7) / 8;
            gemm_bf16_wmma<4><<<gblk, 256, 0, stream>>>(xbf, wbf,      bl[li], xl, ic, hoc, groupsN, waves);
            gemm_bf16_wmma<4><<<gblk, 256, 0, stream>>>(xbf, wbf + nW, br[li], xr, ic, hoc, groupsN, waves);
        } else {                                // layer 4: OC=16, NT=1
            const int waves = NODES / 16;
            const int gblk  = (waves + 7) / 8;
            gemm_bf16_wmma<1><<<gblk, 256, 0, stream>>>(xbf, wbf,      bl[li], xl, ic, hoc, 1, waves);
            gemm_bf16_wmma<1><<<gblk, 256, 0, stream>>>(xbf, wbf + nW, br[li], xr, ic, hoc, 1, waves);
        }

        const int eblk = (ETOT + 7) / 8;
        edge_logits   <<<eblk, 256, 0, stream>>>(xl, xr, at[li], srcF, dstF, logit, nmax, heads, hoc);
        edge_expsum   <<<(ETOT * heads + 255) / 256, 256, 0, stream>>>(dstF, logit, nmax, den, heads);
        edge_aggregate<<<eblk, 256, 0, stream>>>(xl, logit, den, srcF, dstF, accb, heads, hoc);

        if (li < 3) {
            // fused bias + ReLU + f32->bf16: writes next layer's GEMM input directly
            epilogue_relu_bf16<<<(nA + 255) / 256, 256, 0, stream>>>(accb, bo[li], xbf, nA, 255);
        } else {
            epilogue_out<<<(NODES * 16 + 255) / 256, 256, 0, stream>>>(
                accb, bo[li], (float*)d_out, NODES * 16, 15);
        }
    }
}
